// GraphDecoder_82274393522845
// MI455X (gfx1250) — matrix operand order repaired and checked by reading, not compiled, not run on hardware
//
#include <hip/hip_runtime.h>
#include <hip/hip_bf16.h>
#include <math.h>

// Problem constants (match reference)
#define N_   3072
#define IN_  512
#define D_   256
#define H_   8
#define DK_  32
#define OUT_ 512
#define L_   2

typedef __attribute__((ext_vector_type(16))) _Float16 v16h;
typedef __attribute__((ext_vector_type(8)))  _Float16 v8h;
typedef __attribute__((ext_vector_type(2)))  _Float16 v2h;
typedef __attribute__((ext_vector_type(8)))  float    v8f;
typedef __attribute__((ext_vector_type(4)))  float    v4f;
typedef __attribute__((ext_vector_type(2)))  float    v2f;

// ---------------------------------------------------------------------------
// 1) Fold Wq@aq and Wk@ak into Cqk[l][c][d]  (c<8: q-proj scaled 1/sqrt(32),
//    c>=8: k-proj), and zero-fill the 512-float "no bias" vector that
//    immediately follows Cqk in the workspace.
// ---------------------------------------------------------------------------
__global__ void prep_kernel(const float* __restrict__ Wq,
                            const float* __restrict__ Wk,
                            const float* __restrict__ aq,
                            const float* __restrict__ ak,
                            float* __restrict__ Cqk) {
  int gid = blockIdx.x * blockDim.x + threadIdx.x;
  if (gid >= L_ * 16 * D_ + OUT_) return;
  if (gid >= L_ * 16 * D_) {        // zero-bias region
    Cqk[gid] = 0.0f;
    return;
  }
  int l   = gid / (16 * D_);
  int rem = gid - l * 16 * D_;
  int c   = rem >> 8;               // 0..15
  int d   = rem & 255;
  float acc = 0.0f;
  if (c < 8) {
    int h = c;
    #pragma unroll 8
    for (int k = 0; k < DK_; ++k)
      acc += Wq[(size_t)l * D_ * D_ + (size_t)d * D_ + h * DK_ + k] *
             aq[l * H_ * DK_ + h * DK_ + k];
    acc *= 0.17677669529663687f;    // 1/sqrt(32)
  } else {
    int h = c - 8;
    #pragma unroll 8
    for (int k = 0; k < DK_; ++k)
      acc += Wk[(size_t)l * D_ * D_ + (size_t)d * D_ + h * DK_ + k] *
             ak[l * H_ * DK_ + h * DK_ + k];
  }
  Cqk[gid] = acc;
}

// ---------------------------------------------------------------------------
// 2) Tiny matvec: qkattn[c][n] = x[n,:]·Cqk[c,:]  -> [16, N] head-major
// ---------------------------------------------------------------------------
__global__ void qk_kernel(const float* __restrict__ x,
                          const float* __restrict__ Cqk,
                          float* __restrict__ qkattn) {
  int gid = blockIdx.x * blockDim.x + threadIdx.x;
  if (gid >= N_ * 16) return;
  int n = gid >> 4;
  int c = gid & 15;
  const float* xr = x + (size_t)n * D_;
  const float* cr = Cqk + (size_t)c * D_;
  float acc = 0.0f;
  #pragma unroll 4
  for (int d = 0; d < D_; d += 4) {
    v4f xv = *(const v4f*)&xr[d];
    v4f cv = *(const v4f*)&cr[d];
    acc += xv.x * cv.x + xv.y * cv.y + xv.z * cv.z + xv.w * cv.w;
  }
  qkattn[(size_t)c * N_ + n] = acc;
}

// ---------------------------------------------------------------------------
// 3) WMMA f16 GEMM:  C[N_,NC] = A[N_,KD] @ B[KD,NC] + bias[NC]
//    Block = 256 threads = 8 waves; block tile 64x64; wave tile 16x32.
//    A tile row-major [row][K]  -> per-lane two contiguous 8xf16 runs.
//    B tile TRANSPOSED [col][K] -> per-lane two contiguous 8xf16 runs.
//    LDS fills use paired f16 (32-bit DS stores); KD/NC compile-time.
// ---------------------------------------------------------------------------
template <int KD, int NC>
__global__ __launch_bounds__(256) void gemm_kernel(
    const float* __restrict__ A, const float* __restrict__ B,
    const float* __restrict__ bias, float* __restrict__ C) {
  __shared__ _Float16 a_lds[64 * 32];   // [row][K]
  __shared__ _Float16 b_t[64 * 32];     // [col][K]  (transposed)

  const int t    = threadIdx.x;
  const int wave = t >> 5;
  const int lane = t & 31;
  const int wr   = wave >> 1;           // 0..3  (16-row slice)
  const int wc   = wave & 1;            // 0..1  (32-col slice)
  const int row  = lane & 15;
  const int half = lane >> 4;
  const int m0   = blockIdx.y * 64;
  const int n0   = blockIdx.x * 64;

  const int arow  = (wr * 16 + row) * 32;
  const int abase = half ? 8 : 0;
  const int bbase = half ? 8 : 0;
  const int bcol0 = (wc * 32 + row) * 32;
  const int bcol1 = (wc * 32 + 16 + row) * 32;

  v8f acc0 = {};
  v8f acc1 = {};

  for (int kk = 0; kk < KD; kk += 32) {
    __syncthreads();
    #pragma unroll
    for (int q = 0; q < 4; ++q) {
      {                                   // A: pair along K, one b64 load
        int idx2 = q * 512 + t * 2;       // even
        int r = idx2 >> 5, k = idx2 & 31;
        v2f av = *(const v2f*)&A[(size_t)(m0 + r) * KD + kk + k];
        v2h ah; ah.x = (_Float16)av.x; ah.y = (_Float16)av.y;
        *(v2h*)&a_lds[idx2] = ah;
      }
      {                                   // B: pair along K for fixed col
        int p = q * 256 + t;
        int n = p & 63, r2 = (p >> 6) * 2;
        float f0 = B[(size_t)(kk + r2) * NC + n0 + n];
        float f1 = B[(size_t)(kk + r2 + 1) * NC + n0 + n];
        v2h bh; bh.x = (_Float16)f0; bh.y = (_Float16)f1;
        *(v2h*)&b_t[n * 32 + r2] = bh;
      }
    }
    __syncthreads();

    v8h alo = *(const v8h*)&a_lds[arow + abase];
    v8h ahi = *(const v8h*)&a_lds[arow + 16 + abase];
    v16h a = __builtin_shufflevector(alo, ahi, 0, 1, 2, 3, 4, 5, 6, 7,
                                     8, 9, 10, 11, 12, 13, 14, 15);
    v16h b0 = __builtin_shufflevector(*(const v8h*)&b_t[bcol0 + bbase], *(const v8h*)&b_t[bcol0 + 16 + bbase], 0, 1, 2, 3, 4, 5, 6, 7, 8, 9, 10, 11, 12, 13, 14, 15);
    v16h b1 = __builtin_shufflevector(*(const v8h*)&b_t[bcol1 + bbase], *(const v8h*)&b_t[bcol1 + 16 + bbase], 0, 1, 2, 3, 4, 5, 6, 7, 8, 9, 10, 11, 12, 13, 14, 15);

    acc0 = __builtin_amdgcn_wmma_f32_16x16x32_f16(false, a, false, b0,
                                                  (short)0, acc0, false, false);
    acc1 = __builtin_amdgcn_wmma_f32_16x16x32_f16(false, a, false, b1,
                                                  (short)0, acc1, false, false);
  }

  const int gc0 = n0 + wc * 32 + row;
  const float b0v = bias[gc0];
  const float b1v = bias[gc0 + 16];
  float* crow = C + (size_t)(m0 + wr * 16 + (half ? 8 : 0)) * NC + gc0;
  #pragma unroll
  for (int r = 0; r < 8; ++r) {
    crow[(size_t)r * NC]      = acc0[r] + b0v;
    crow[(size_t)r * NC + 16] = acc1[r] + b1v;
  }
}

// ---------------------------------------------------------------------------
// 4) Fused GAT attention for one 16-row i-tile, all 8 heads (1 wave/head).
//    scores[h,i,j] = leaky_relu(mask[i,j]*(q[h,i]+k[h,j])), never materialized.
//    Pass A: streaming max/sum per (h,row): tile-local max, then one
//            rescale + 16 exps per tile (no serial exp chain).
//    Pass B: P = exp(s - m) built directly in the WMMA A-fragment layout
//            (lane == row -> per-lane scalars for m, 1/sum), V staged in LDS
//            transposed [col][K] for contiguous B-fragment loads,
//            accumulate P@V with v_wmma_f32_16x16x32_f16 into f32.
// ---------------------------------------------------------------------------
__global__ __launch_bounds__(256) void attn_kernel(
    const float* __restrict__ qkattn,   // [16, N]: rows 0..7 q, 8..15 k
    const float* __restrict__ mask,     // [N, N]
    const float* __restrict__ V,        // [N, D]
    float* __restrict__ xout) {         // [N, D]
  __shared__ float    mask_lds[16 * 32];   // [row][j]
  __shared__ float    kattn_lds[H_ * 32];  // [head][j]
  __shared__ _Float16 v_t[D_ * 32];        // [col][j]  (transposed)
  __shared__ float    pm[H_][2][16];
  __shared__ float    ps[H_][2][16];
  __shared__ float    row_m[H_][16];
  __shared__ float    row_is[H_][16];

  const int t    = threadIdx.x;
  const int h    = t >> 5;        // wave == head
  const int lane = t & 31;
  const int row  = lane & 15;
  const int half = lane >> 4;
  const int i0   = blockIdx.x * 16;

  const float q_i = qkattn[(size_t)h * N_ + i0 + row];

  // ---------------- Pass A: row max & sum (streaming) ----------------
  float m = -1e30f, s = 0.0f;
  for (int j0 = 0; j0 < N_; j0 += 32) {
    __syncthreads();
    {                                       // mask tile: paired b64 loads
      int idx2 = t * 2;
      int r = idx2 >> 5, jj = idx2 & 31;
      *(v2f*)&mask_lds[idx2] = *(const v2f*)&mask[(size_t)(i0 + r) * N_ + j0 + jj];
    }
    { int c = t >> 5, jj = t & 31;
      kattn_lds[t] = qkattn[(size_t)(8 + c) * N_ + j0 + jj]; }
    __syncthreads();

    float sc[16];                           // contiguous 16-j slice
    #pragma unroll
    for (int q = 0; q < 4; ++q) {
      v4f mk = *(const v4f*)&mask_lds[row * 32 + half * 16 + q * 4];
      v4f kj = *(const v4f*)&kattn_lds[h * 32 + half * 16 + q * 4];
      #pragma unroll
      for (int e = 0; e < 4; ++e) {
        float v = mk[e] * (q_i + kj[e]);
        sc[q * 4 + e] = v > 0.0f ? v : 0.2f * v;   // leaky_relu(0.2)
      }
    }
    float tm = sc[0];
    #pragma unroll
    for (int e = 1; e < 16; ++e) tm = fmaxf(tm, sc[e]);
    float mn = fmaxf(m, tm);
    float acc = 0.0f;
    #pragma unroll
    for (int e = 0; e < 16; ++e) acc += __expf(sc[e] - mn);
    s = s * __expf(m - mn) + acc;
    m = mn;
  }
  pm[h][half][row] = m;
  ps[h][half][row] = s;
  __syncthreads();
  if (half == 0) {
    float m0 = pm[h][0][row], m1 = pm[h][1][row];
    float mt = fmaxf(m0, m1);
    float st = ps[h][0][row] * __expf(m0 - mt) + ps[h][1][row] * __expf(m1 - mt);
    row_m[h][row]  = mt;
    row_is[h][row] = 1.0f / st;
  }
  __syncthreads();

  // ---------------- Pass B: out = softmax(P) @ V via WMMA ----------------
  const float m_i   = row_m[h][row];
  const int   kb    = half ? 8 : 0;        // A-frag K base
  const int   bb    = half ? 8 : 0;        // B-frag K base
  const int   bcol0 = (h * DK_ + row) * 32;
  const int   bcol1 = (h * DK_ + 16 + row) * 32;

  v8f acc0 = {};
  v8f acc1 = {};
  for (int j0 = 0; j0 < N_; j0 += 32) {
    __syncthreads();
    {
      int idx2 = t * 2;
      int r = idx2 >> 5, jj = idx2 & 31;
      *(v2f*)&mask_lds[idx2] = *(const v2f*)&mask[(size_t)(i0 + r) * N_ + j0 + jj];
    }
    { int c = t >> 5, jj = t & 31;
      kattn_lds[t] = qkattn[(size_t)(8 + c) * N_ + j0 + jj]; }
    #pragma unroll 8
    for (int k2 = 0; k2 < 32; k2 += 2) {    // V tile -> transposed f16 [col][k]
      float f0 = V[(size_t)(j0 + k2) * D_ + t];
      float f1 = V[(size_t)(j0 + k2 + 1) * D_ + t];
      v2h p; p.x = (_Float16)f0; p.y = (_Float16)f1;
      *(v2h*)&v_t[t * 32 + k2] = p;
    }
    __syncthreads();

    // A-frag K pattern: e=0..7 -> kb+e ; e=8..15 -> kb+16+(e-8)
    float mkv[16], kjv[16];
    #pragma unroll
    for (int q = 0; q < 2; ++q) {
      *(v4f*)&mkv[q * 4]     = *(const v4f*)&mask_lds[row * 32 + kb + q * 4];
      *(v4f*)&mkv[8 + q * 4] = *(const v4f*)&mask_lds[row * 32 + kb + 16 + q * 4];
      *(v4f*)&kjv[q * 4]     = *(const v4f*)&kattn_lds[h * 32 + kb + q * 4];
      *(v4f*)&kjv[8 + q * 4] = *(const v4f*)&kattn_lds[h * 32 + kb + 16 + q * 4];
    }
    v16h a;
    #pragma unroll
    for (int e = 0; e < 16; ++e) {
      float v = mkv[e] * (q_i + kjv[e]);
      v = v > 0.0f ? v : 0.2f * v;
      a[e] = (_Float16)__expf(v - m_i);     // softmax numerator
    }
    v16h b0 = __builtin_shufflevector(*(const v8h*)&v_t[bcol0 + bb], *(const v8h*)&v_t[bcol0 + 16 + bb], 0, 1, 2, 3, 4, 5, 6, 7, 8, 9, 10, 11, 12, 13, 14, 15);
    v16h b1 = __builtin_shufflevector(*(const v8h*)&v_t[bcol1 + bb], *(const v8h*)&v_t[bcol1 + 16 + bb], 0, 1, 2, 3, 4, 5, 6, 7, 8, 9, 10, 11, 12, 13, 14, 15);

    acc0 = __builtin_amdgcn_wmma_f32_16x16x32_f16(false, a, false, b0,
                                                  (short)0, acc0, false, false);
    acc1 = __builtin_amdgcn_wmma_f32_16x16x32_f16(false, a, false, b1,
                                                  (short)0, acc1, false, false);
  }

  const float* isrow = &row_is[h][half ? 8 : 0];
  float* obase = xout + (size_t)(i0 + (half ? 8 : 0)) * D_ + h * DK_ + row;
  #pragma unroll
  for (int r = 0; r < 8; ++r) {
    float is = isrow[r];
    obase[(size_t)r * D_]      = acc0[r] * is;
    obase[(size_t)r * D_ + 16] = acc1[r] * is;
  }
}

// ---------------------------------------------------------------------------
// Orchestration
// ---------------------------------------------------------------------------
extern "C" void kernel_launch(void* const* d_in, const int* in_sizes, int n_in,
                              void* d_out, int out_size, void* d_ws, size_t ws_size,
                              hipStream_t stream) {
  (void)in_sizes; (void)n_in; (void)out_size; (void)ws_size;

  const float* noise = (const float*)d_in[0];
  // d_in[1] (resist_enc_embed) and d_in[3] (cross_mask) are unused by reference
  const float* mask  = (const float*)d_in[2];
  const float* w_in  = (const float*)d_in[4];
  const float* b_in  = (const float*)d_in[5];
  const float* Wq    = (const float*)d_in[6];
  const float* Wk    = (const float*)d_in[7];
  const float* Wv    = (const float*)d_in[8];
  const float* aq    = (const float*)d_in[9];
  const float* ak    = (const float*)d_in[10];
  const float* w_out = (const float*)d_in[11];
  const float* b_out = (const float*)d_in[12];
  float* out = (float*)d_out;

  float* ws    = (float*)d_ws;
  float* x0    = ws;                        // [N, D]
  float* x1    = x0 + (size_t)N_ * D_;      // [N, D]
  float* Vb    = x1 + (size_t)N_ * D_;      // [N, D]
  float* qk    = Vb + (size_t)N_ * D_;      // [16, N]
  float* Cqk   = qk + (size_t)16 * N_;      // [L, 16, D]
  float* zerob = Cqk + (size_t)L_ * 16 * D_;// [OUT_] zeros (bias for no-bias)

  // Fold attention vectors into per-layer projections + zero-bias fill
  prep_kernel<<<(L_ * 16 * D_ + OUT_ + 255) / 256, 256, 0, stream>>>(
      Wq, Wk, aq, ak, Cqk);

  // x = noise @ w_in + b_in       [3072,512]x[512,256]
  gemm_kernel<IN_, D_><<<dim3(D_ / 64, N_ / 64), 256, 0, stream>>>(
      noise, w_in, b_in, x0);

  float* xin  = x0;
  float* xnew = x1;
  for (int l = 0; l < L_; ++l) {
    // V = x @ Wv[l]               [3072,256]x[256,256]
    gemm_kernel<D_, D_><<<dim3(D_ / 64, N_ / 64), 256, 0, stream>>>(
        xin, Wv + (size_t)l * D_ * D_, zerob, Vb);
    // q_attn / k_attn
    qk_kernel<<<(N_ * 16 + 255) / 256, 256, 0, stream>>>(
        xin, Cqk + (size_t)l * 16 * D_, qk);
    // fused masked-GAT softmax + P@V
    attn_kernel<<<N_ / 16, 256, 0, stream>>>(qk, mask, Vb, xnew);
    float* tmp = xin; xin = xnew; xnew = tmp;
  }

  // out = x @ w_out + b_out       [3072,256]x[256,512]
  gemm_kernel<D_, OUT_><<<dim3(OUT_ / 64, N_ / 64), 256, 0, stream>>>(
      xin, w_out, b_out, out);
}
